// DynamicGraphBuilder_78546361909463
// MI455X (gfx1250) — matrix operand order repaired and checked by reading, not compiled, not run on hardware
//
#include <hip/hip_runtime.h>
#include <hip/hip_bf16.h>

// ---------------------------------------------------------------------------
// DynamicGraphBuilder for MI455X (gfx1250, wave32, WMMA).
// sim GEMM: v_wmma_f32_16x16x32_f16, two independent accumulator chains,
// all 8 fragment loads clause'd ahead of the WMMAs (sched_barrier) so one
// s_wait_loadcnt covers the whole tile. Operands f16, L2-resident (2 MB).
// edge_attr (256 MB) produced in one fused pass with nontemporal stores =
// HBM floor (~11us @ 23.3 TB/s). Haversine is trig-free per pair (per-node
// cos/sin precompute -> FMAs + asin(sqrt)). Per-node mask tables staged
// per-block into LDS as 16-bit masks.
// ---------------------------------------------------------------------------

#define N_NODES 8192
#define D_FEAT  128
#define N_EDGES 65536

typedef __attribute__((ext_vector_type(16))) _Float16 v16h;
typedef __attribute__((ext_vector_type(8)))  _Float16 v8h;
typedef __attribute__((ext_vector_type(8)))  float    v8f;

union FragU { v16h v; v8h h8[2]; };

// ------------------------- hash PRNG (stand-in for threefry) ---------------
#define SEED_KF 0xA511E9B3u
#define SEED_KG 0xC2B2AE35u
#define SEED_KS 0x27D4EB2Fu
#define SEED_KA 0x165667B1u

__device__ __forceinline__ unsigned hmix(unsigned h) {
    h ^= h >> 16; h *= 0x7FEB352Du;
    h ^= h >> 15; h *= 0x846CA68Bu;
    h ^= h >> 16; return h;
}
__device__ __forceinline__ unsigned hash2(unsigned salt, unsigned a, unsigned b) {
    return hmix(salt ^ (a * 0x85EBCA77u) ^ (b * 0xC2B2AE3Du));
}
__device__ __forceinline__ float u01(unsigned h) {
    return (float)(h >> 8) * (1.0f / 16777216.0f);
}

// ------------------------- WMMA fragment loaders ---------------------------
// A: 16x32 f16, lane<16 holds row M=lane, K = {k0..k0+7, k0+16..k0+23};
//    lane>=16 holds row M=lane-16, K = {k0+8..k0+15, k0+24..k0+31}.
__device__ __forceinline__ v16h load_a_frag(const _Float16* xn, int row0, int k0, int lane) {
    int m  = lane & 15;
    int hi = lane >> 4;
    const _Float16* p = xn + ((size_t)(row0 + m) << 7) + k0 + (hi ? 8 : 0);
    FragU f;
    f.h8[0] = *(const v8h*)p;
    f.h8[1] = *(const v8h*)(p + 16);
    return f.v;
}
// B: 32x16 f16, lane<16 holds col N=lane, K = {k0..k0+7, k0+16..k0+23};
//    lane>=16 holds col N=lane-16, K = {k0+8..k0+15, k0+24..k0+31}.  B[k][n] = xn[n][k].
__device__ __forceinline__ v16h load_b_frag(const _Float16* xn, int col0, int k0, int lane) {
    int n  = lane & 15;
    int hi = lane >> 4;
    const _Float16* p = xn + ((size_t)(col0 + n) << 7) + k0 + (hi ? 8 : 0);
    FragU f;
    f.h8[0] = *(const v8h*)p;
    f.h8[1] = *(const v8h*)(p + 16);
    return f.v;
}

// 16x16 sim tile over K=128: all 8 fragment loads issued first (kept ahead of
// the math by a scheduling barrier -> single s_wait_loadcnt), then two
// independent WMMA chains (c0: K0,K2 / c1: K1,K3) merged at the end.
__device__ __forceinline__ v8f sim_tile(const _Float16* xn, int row0, int col0, int lane) {
    v16h a[4], b[4];
#pragma unroll
    for (int kk = 0; kk < 4; kk++) {
        a[kk] = load_a_frag(xn, row0, kk * 32, lane);
        b[kk] = load_b_frag(xn, col0, kk * 32, lane);
    }
    __builtin_amdgcn_sched_barrier(0);   // keep all loads ahead of the WMMAs
    v8f c0 = {}, c1 = {};
    c0 = __builtin_amdgcn_wmma_f32_16x16x32_f16(false, a[0], false, b[0], (short)0, c0, false, false);
    c1 = __builtin_amdgcn_wmma_f32_16x16x32_f16(false, a[1], false, b[1], (short)0, c1, false, false);
    c0 = __builtin_amdgcn_wmma_f32_16x16x32_f16(false, a[2], false, b[2], (short)0, c0, false, false);
    c1 = __builtin_amdgcn_wmma_f32_16x16x32_f16(false, a[3], false, b[3], (short)0, c1, false, false);
    return c0 + c1;
}

// 9-element register insertion (fully unrolled, stays in VGPRs)
#define INSERT9(BV, BI, VAL, IDX)                                              \
    if ((VAL) > BV[8]) {                                                       \
        BV[8] = (VAL); BI[8] = (IDX);                                          \
        _Pragma("unroll")                                                      \
        for (int _s = 8; _s > 0; _s--) {                                       \
            if (BV[_s] > BV[_s - 1]) {                                         \
                float _tv = BV[_s]; BV[_s] = BV[_s - 1]; BV[_s - 1] = _tv;     \
                int _ti = BI[_s]; BI[_s] = BI[_s - 1]; BI[_s - 1] = _ti;       \
            }                                                                  \
        }                                                                      \
    }

// ------------------------- prep kernels ------------------------------------
__global__ __launch_bounds__(256) void zero_adj_kernel(unsigned* adjw) {
    int t = blockIdx.x * blockDim.x + threadIdx.x;      // 524288 threads
#pragma unroll
    for (int k = 0; k < 4; k++) adjw[t + k * 524288] = 0u;
}

// per-node (cos lat, sin lat, cos lon, sin lon) -> trig-free pairwise haversine
__global__ __launch_bounds__(256) void geo_prep_kernel(const float* coords, float4* geo4) {
    int i = blockIdx.x * blockDim.x + threadIdx.x;
    if (i < N_NODES) {
        const float D2R = 0.017453292519943295f;
        float lo = coords[2 * i + 0] * D2R;
        float la = coords[2 * i + 1] * D2R;
        float4 g;
        g.x = __cosf(la); g.y = __sinf(la);
        g.z = __cosf(lo); g.w = __sinf(lo);
        geo4[i] = g;
    }
}

__global__ __launch_bounds__(256) void normalize_kernel(const float* x, _Float16* xn) {
    int lane = threadIdx.x & 31;
    int row  = blockIdx.x * 8 + (threadIdx.x >> 5);
    const float4* xr = (const float4*)(x + ((size_t)row << 7));
    float4 v = xr[lane];
    float ss = v.x * v.x + v.y * v.y + v.z * v.z + v.w * v.w;
#pragma unroll
    for (int m = 16; m > 0; m >>= 1) ss += __shfl_xor(ss, m, 32);
    float scale = 1.0f / fmaxf(sqrtf(ss), 1e-12f);
    struct H4 { _Float16 a, b, c, d; } o;
    o.a = (_Float16)(v.x * scale); o.b = (_Float16)(v.y * scale);
    o.c = (_Float16)(v.z * scale); o.d = (_Float16)(v.w * scale);
    ((H4*)(xn + ((size_t)row << 7)))[lane] = o;
}

__global__ __launch_bounds__(256) void scatter_adj_kernel(const int* ei, unsigned* adjw) {
    int e = blockIdx.x * blockDim.x + threadIdx.x;
    if (e < N_EDGES) {
        int r = ei[e];
        int c = ei[N_EDGES + e];
        atomicOr(&adjw[((size_t)r << 8) + (c >> 5)], 1u << (c & 31));
    }
}

// ------------------------- per-row top-9 via WMMA --------------------------
__global__ __launch_bounds__(256) void topk_kernel(const _Float16* __restrict__ xn,
                                                   int* __restrict__ topk) {
    __shared__ float simLDS[8][16][16];          // 8 KB
    __shared__ float mergeV[128][9];             // 4.5 KB
    __shared__ int   mergeI[128][9];             // 4.5 KB
    int lane = threadIdx.x & 31;
    int wave = threadIdx.x >> 5;
    int tid  = threadIdx.x;
    int row0 = blockIdx.x * 16;

    float bestv[9]; int besti[9];
#pragma unroll
    for (int s = 0; s < 9; s++) { bestv[s] = -1e30f; besti[s] = 0; }

    int selR = tid & 15, selW = tid >> 4;        // valid when tid < 128

    for (int jt0 = 0; jt0 < N_NODES / 16; jt0 += 8) {
        __builtin_prefetch(xn + ((size_t)(((jt0 + 8) & 511) * 16) << 7), 0, 3);

        v8f c = sim_tile(xn, row0, (jt0 + wave) * 16, lane);
        int hi = (lane >> 4) << 3;
#pragma unroll
        for (int v = 0; v < 8; v++) simLDS[wave][v + hi][lane & 15] = c[v];
        __syncthreads();

        if (tid < 128) {
            int jbase = (jt0 + selW) * 16;
#pragma unroll
            for (int cc = 0; cc < 16; cc++) {
                float val = simLDS[selW][selR][cc];
                INSERT9(bestv, besti, val, jbase + cc);
            }
        }
        __syncthreads();
    }

    if (tid < 128) {
#pragma unroll
        for (int s = 0; s < 9; s++) { mergeV[tid][s] = bestv[s]; mergeI[tid][s] = besti[s]; }
    }
    __syncthreads();
    if (tid < 16) {
        for (int w = 1; w < 8; w++) {
            int t2 = tid + (w << 4);
#pragma unroll
            for (int s = 0; s < 9; s++) {
                float val = mergeV[t2][s];
                INSERT9(bestv, besti, val, mergeI[t2][s]);
            }
        }
#pragma unroll
        for (int s = 0; s < 9; s++) topk[(row0 + tid) * 9 + s] = besti[s];
    }
}

// ------------------------- fused output pass -------------------------------
__global__ __launch_bounds__(256) void edge_attr_kernel(
        const _Float16* __restrict__ xn, const unsigned* __restrict__ adjw,
        const int* __restrict__ topk, const float4* __restrict__ geo4,
        const float* fwp, const float* gwp, const float* glwp,
        float* __restrict__ out) {
    __shared__ float4         S_geo_r[16];       // (cl, sl, co, so) per row node
    __shared__ float4         S_geo_c[128];
    __shared__ unsigned short S_adjrow[16][8], S_knnrow[16][8], S_globrow[16][8];
    __shared__ unsigned short S_adjcol[128],  S_knncol[128],  S_globcol[128];

    int lane = threadIdx.x & 31;
    int wave = threadIdx.x >> 5;
    int tid  = threadIdx.x;
    int row0     = blockIdx.x * 16;
    int colbase0 = blockIdx.y * 128;

    float fw = *fwp, gw = *gwp, glw = *glwp;

    // ---------------- phase 1: cooperative staging ----------------
    if (tid < 128) {
        int r  = tid & 15, ct = tid >> 4;
        int i  = row0 + r;
        int c0 = colbase0 + ct * 16;
        unsigned w32 = adjw[((size_t)i << 8) + (c0 >> 5)];
        S_adjrow[r][ct] = (unsigned short)((w32 >> (c0 & 16)) & 0xFFFFu);

        unsigned km = 0, gm = 0;
#pragma unroll
        for (int s = 0; s < 8; s++) {
            int t = topk[i * 9 + 1 + s];
            bool acc = u01(hash2(SEED_KF, (unsigned)i, (unsigned)s)) < fw;
            unsigned d = (unsigned)(t - c0);
            km |= (acc && (t != i) && d < 16u) ? (1u << d) : 0u;
        }
#pragma unroll
        for (int s = 0; s < 3; s++) {
            int off = 1 + (int)(hash2(SEED_KS, (unsigned)i, (unsigned)s) % (N_NODES - 1));
            int g   = (i + off) & (N_NODES - 1);
            bool acc = u01(hash2(SEED_KA, (unsigned)i, (unsigned)s)) < glw;
            unsigned d = (unsigned)(g - c0);
            gm |= (acc && d < 16u) ? (1u << d) : 0u;
        }
        S_knnrow[r][ct]  = (unsigned short)km;
        S_globrow[r][ct] = (unsigned short)gm;

        if (tid < 16) S_geo_r[tid] = geo4[i];
    } else {
        int jl = tid - 128;
        int j  = colbase0 + jl;
        unsigned w32 = adjw[((size_t)j << 8) + (row0 >> 5)];
        S_adjcol[jl] = (unsigned short)((w32 >> (row0 & 16)) & 0xFFFFu);

        unsigned km = 0, gm = 0;
#pragma unroll
        for (int s = 0; s < 8; s++) {
            int t = topk[j * 9 + 1 + s];
            bool acc = u01(hash2(SEED_KF, (unsigned)j, (unsigned)s)) < fw;
            unsigned d = (unsigned)(t - row0);
            km |= (acc && (t != j) && d < 16u) ? (1u << d) : 0u;
        }
#pragma unroll
        for (int s = 0; s < 3; s++) {
            int off = 1 + (int)(hash2(SEED_KS, (unsigned)j, (unsigned)s) % (N_NODES - 1));
            int g   = (j + off) & (N_NODES - 1);
            bool acc = u01(hash2(SEED_KA, (unsigned)j, (unsigned)s)) < glw;
            unsigned d = (unsigned)(g - row0);
            gm |= (acc && d < 16u) ? (1u << d) : 0u;
        }
        S_knncol[jl]  = (unsigned short)km;
        S_globcol[jl] = (unsigned short)gm;
        S_geo_c[jl] = geo4[j];
    }
    __syncthreads();

    // ---------------- phase 2: WMMA tile + fused tail ----------------
    int ct   = wave;
    int col0 = colbase0 + ct * 16;
    v8f c = sim_tile(xn, row0, col0, lane);

    int nloc = lane & 15;             // column within tile
    int jl   = ct * 16 + nloc;
    int j    = colbase0 + jl;
    float4 gj = S_geo_c[jl];
    unsigned adjc = S_adjcol[jl], knnc = S_knncol[jl], glbc = S_globcol[jl];
    int hi8 = (lane >> 4) << 3;

#pragma unroll
    for (int v = 0; v < 8; v++) {
        int rloc = v + hi8;
        int i    = row0 + rloc;
        float sim = c[v];

        unsigned rowbits = ((unsigned)S_adjrow[rloc][ct])
                         | ((unsigned)S_knnrow[rloc][ct] << 16);
        bool adj_ij = (rowbits >> nloc) & 1u;
        bool knn_ij = (rowbits >> (16 + nloc)) & 1u;
        bool glb_ij = (S_globrow[rloc][ct] >> nloc) & 1u;
        bool adj_ji = (adjc >> rloc) & 1u;
        bool knn_ji = (knnc >> rloc) & 1u;
        bool glb_ji = (glbc >> rloc) & 1u;

        // trig-free haversine: sin^2(d/2) = (1 - cos d)/2 with
        // cos d from angle-difference identity on precomputed cos/sin
        float4 gi = S_geo_r[rloc];
        float pc      = gi.x * gj.x;                 // cos(lat_i)cos(lat_j)
        float cosdlat = pc + gi.y * gj.y;
        float cosdlon = gi.z * gj.z + gi.w * gj.w;
        float hav = 0.5f * (1.0f - cosdlat) + pc * 0.5f * (1.0f - cosdlon);
        hav = fminf(fmaxf(hav, 0.0f), 1.0f);
        float dist  = 2.0f * 6371.0f * asinf(sqrtf(hav));
        float gterm = gw / (1.0f + dist);            // == geo_prob

        bool neq = (i != j);
        bool geo_ij = (u01(hash2(SEED_KG, (unsigned)i, (unsigned)j)) < gterm) && neq;
        bool geo_ji = (u01(hash2(SEED_KG, (unsigned)j, (unsigned)i)) < gterm) && neq;

        bool af = adj_ij | knn_ij | geo_ij | glb_ij |
                  adj_ji | knn_ji | geo_ji | glb_ji | !neq;

        float attr = af ? (fw * sim + gterm) : 0.0f;
        // pure streaming output: nontemporal store (TH_NT), keep L2 for tables
        __builtin_nontemporal_store(attr, out + (((size_t)i << 13) + j));
    }
}

// ------------------------- launch ------------------------------------------
extern "C" void kernel_launch(void* const* d_in, const int* in_sizes, int n_in,
                              void* d_out, int out_size, void* d_ws, size_t ws_size,
                              hipStream_t stream) {
    const float* x      = (const float*)d_in[0];
    const float* coords = (const float*)d_in[1];
    const float* fwp    = (const float*)d_in[2];
    const float* gwp    = (const float*)d_in[3];
    const float* glwp   = (const float*)d_in[4];
    const int*   ei     = (const int*)d_in[5];
    float*       out    = (float*)d_out;

    // workspace carve-out (~10.5 MB)
    char* ws = (char*)d_ws;
    _Float16* xn   = (_Float16*)ws;                                  // 2 MB
    unsigned* adjw = (unsigned*)(ws + (2u << 20));                   // 8 MB bitmask
    float4*   geo4 = (float4*)(ws + (10u << 20));                    // 128 KB
    int*      topk = (int*)(ws + (10u << 20) + (N_NODES * 16));      // 288 KB

    zero_adj_kernel   <<<2048, 256, 0, stream>>>(adjw);
    geo_prep_kernel   <<<N_NODES / 256, 256, 0, stream>>>(coords, geo4);
    normalize_kernel  <<<N_NODES / 8, 256, 0, stream>>>(x, xn);
    scatter_adj_kernel<<<N_EDGES / 256, 256, 0, stream>>>(ei, adjw);
    topk_kernel       <<<N_NODES / 16, 256, 0, stream>>>(xn, topk);
    edge_attr_kernel  <<<dim3(N_NODES / 16, N_NODES / (16 * 8)), 256, 0, stream>>>(
        xn, adjw, topk, geo4, fwp, gwp, glwp, out);
}
